// Gemma3_86157043958052
// MI455X (gfx1250) — matrix operand order repaired and checked by reading, not compiled, not run on hardware
//
#include <hip/hip_runtime.h>
#include <cstddef>

// ---------------------------------------------------------------------------
// Gemma-3 sliding-window attention block for MI455X (gfx1250, wave32, WMMA).
// Matmuls run on v_wmma_f32_16x16x32_bf16 (fp32 accumulate); norms / RoPE /
// softmax run in fp32 VALU.  ~100 GFLOP total, HBM-light (~200MB), so the
// tensor path is the only lever that matters on this chip.
// Global->LDS staging uses GLOBAL_LOAD_ASYNC_TO_LDS_B128 (ASYNCcnt) when the
// toolchain exposes the builtin, with a register-staged fallback otherwise.
// ---------------------------------------------------------------------------

typedef __attribute__((ext_vector_type(16))) __bf16 v16bf;
typedef __attribute__((ext_vector_type(8)))  __bf16 v8bf;
typedef __attribute__((ext_vector_type(8)))  float  v8f;
typedef __attribute__((ext_vector_type(4)))  int    v4i;

#if defined(__AMDGCN__) && __has_builtin(__builtin_amdgcn_global_load_async_to_lds_b128)
  #define HAVE_ASYNC_LDS 1
#else
  #define HAVE_ASYNC_LDS 0
#endif

static __device__ inline __bf16 f2bf(float f) {
  unsigned u = __builtin_bit_cast(unsigned, f);
  unsigned r = u + 0x7FFFu + ((u >> 16) & 1u);   // round-to-nearest-even
  unsigned short h = (unsigned short)(r >> 16);
  return __builtin_bit_cast(__bf16, h);
}

static __device__ inline v16bf cat16(v8bf lo, v8bf hi) {
  return __builtin_shufflevector(lo, hi, 0,1,2,3,4,5,6,7,8,9,10,11,12,13,14,15);
}

// 16-byte global->LDS copy: async DMA path on gfx1250, VGPR-staged fallback.
static __device__ inline void g2l_copy16(const __bf16* g, __bf16* l) {
#if HAVE_ASYNC_LDS
  __builtin_amdgcn_global_load_async_to_lds_b128(
      (__attribute__((address_space(1))) v4i*)g,
      (__attribute__((address_space(3))) v4i*)l, 0, 0);
#else
  *(v8bf*)l = *(const v8bf*)g;
#endif
}

static __device__ inline void g2l_wait() {
#if HAVE_ASYNC_LDS
  #if __has_builtin(__builtin_amdgcn_s_wait_asynccnt)
  __builtin_amdgcn_s_wait_asynccnt(0);
  #else
  asm volatile("s_wait_asynccnt 0" ::: "memory");
  #endif
#endif
}

// ---------------------------------------------------------------------------
// elementwise f32 -> bf16
// ---------------------------------------------------------------------------
__global__ __launch_bounds__(256)
void cvt_f32_bf16(const float* __restrict__ in, __bf16* __restrict__ out, int n) {
  int i = blockIdx.x * 256 + threadIdx.x;
  if (i < n) out[i] = f2bf(in[i]);
}

// ---------------------------------------------------------------------------
// W [N,K] f32 row-major  ->  WT [K,N] bf16 row-major (tiled via LDS)
// ---------------------------------------------------------------------------
__global__ __launch_bounds__(256)
void transpose_cvt(const float* __restrict__ W, __bf16* __restrict__ WT,
                   int N, int K) {
  __shared__ __bf16 tile[32][33];
  int nb = blockIdx.y * 32, kb = blockIdx.x * 32;
  int tx = threadIdx.x & 31, ty = threadIdx.x >> 5;   // 32 x 8
  #pragma unroll
  for (int i = 0; i < 32; i += 8)
    tile[ty + i][tx] = f2bf(W[(size_t)(nb + ty + i) * K + kb + tx]);
  __syncthreads();
  #pragma unroll
  for (int i = 0; i < 32; i += 8)
    WT[(size_t)(kb + ty + i) * N + nb + tx] = tile[tx][ty + i];
}

// ---------------------------------------------------------------------------
// C[M,N] f32 = A[M,K] bf16 (row-major) x Bt[K,N] bf16 (row-major, = W^T)
// BM=128 BN=128 BK=32, 256 threads = 8 waves (4 in M x 2 in N),
// each wave: 32x64 tile = 2x4 WMMA accumulators.
// ---------------------------------------------------------------------------
#define BM 128
#define BN 128
#define BK 32

__global__ __launch_bounds__(256)
void gemm_bf16_f32(const __bf16* __restrict__ A, const __bf16* __restrict__ Bt,
                   float* __restrict__ C, int M, int N, int K) {
  __shared__ __bf16 As[BM][BK];   // [row][k]   (k contiguous)
  __shared__ __bf16 Bs[BN][BK];   // [col][k]   (k contiguous)

  const int tid  = threadIdx.x;
  const int lane = tid & 31;
  const int wave = tid >> 5;
  const int wm   = wave & 3;           // row group (32 rows)
  const int wn   = wave >> 2;          // col group (64 cols)
  const int rowBase = blockIdx.y * BM;
  const int colBase = blockIdx.x * BN;

  const int l16   = lane & 15;
  const int aoff  = (lane < 16) ? 0 : 16;   // A-fragment K half
  const int koff  = (lane < 16) ? 0 : 16;   // B-fragment K half
  const int half8 = (lane < 16) ? 0 : 8;    // C/D row offset

  const v8f vzero = {0.f,0.f,0.f,0.f,0.f,0.f,0.f,0.f};
  v8f acc[2][4];
  #pragma unroll
  for (int mi = 0; mi < 2; ++mi)
    #pragma unroll
    for (int ni = 0; ni < 4; ++ni) acc[mi][ni] = vzero;

  for (int k0 = 0; k0 < K; k0 += BK) {
    // stage A tile: 512 x 16B chunks, 2 per thread (async DMA when available)
    #pragma unroll
    for (int i = 0; i < 2; ++i) {
      int chunk = tid * 2 + i;                  // 0..511
      int r  = chunk >> 2;
      int kc = (chunk & 3) * 8;
      g2l_copy16(&A[(size_t)(rowBase + r) * K + k0 + kc], &As[r][kc]);
    }
    // stage B tile (transpose k-major into Bs[col][k]) -- must go through VGPRs
    #pragma unroll
    for (int i = 0; i < 2; ++i) {
      int chunk = tid * 2 + i;                  // 0..511
      int kk = chunk >> 4;
      int nc = (chunk & 15) * 8;
      v8bf tmp = *(const v8bf*)&Bt[(size_t)(k0 + kk) * N + colBase + nc];
      #pragma unroll
      for (int j = 0; j < 8; ++j) Bs[nc + j][kk] = tmp[j];
    }
    if (k0 + BK < K) {   // gfx1250 global_prefetch_b8 for next tiles
      __builtin_prefetch(&A[(size_t)(rowBase + (tid >> 1)) * K + k0 + BK], 0, 1);
      __builtin_prefetch(&Bt[(size_t)(k0 + BK + (tid >> 4)) * N + colBase], 0, 1);
    }
    g2l_wait();          // ASYNCcnt == 0 before the workgroup barrier
    __syncthreads();

    v16bf af[2], bfr[4];
    #pragma unroll
    for (int mi = 0; mi < 2; ++mi) {
      int r = wm * 32 + mi * 16 + l16;
      af[mi] = *(const v16bf*)&As[r][aoff];
    }
    #pragma unroll
    for (int ni = 0; ni < 4; ++ni) {
      int c = wn * 64 + ni * 16 + l16;
      bfr[ni] = *(const v16bf*)&Bs[c][koff];
    }
    #pragma unroll
    for (int mi = 0; mi < 2; ++mi)
      #pragma unroll
      for (int ni = 0; ni < 4; ++ni)
        acc[mi][ni] = __builtin_amdgcn_wmma_f32_16x16x32_bf16(
            false, af[mi], false, bfr[ni], (short)0, acc[mi][ni], false, false);
    __syncthreads();
  }

  #pragma unroll
  for (int mi = 0; mi < 2; ++mi) {
    #pragma unroll
    for (int ni = 0; ni < 4; ++ni) {
      int r0 = rowBase + wm * 32 + mi * 16 + half8;
      int c  = colBase + wn * 64 + ni * 16 + l16;
      #pragma unroll
      for (int v = 0; v < 8; ++v)
        C[(size_t)(r0 + v) * N + c] = acc[mi][ni][v];
    }
  }
}

// ---------------------------------------------------------------------------
// Gemma RMSNorm (scale = 1+w) + RoPE (+ optional q-scale), f32 in, bf16 out.
// One 128-thread block per (token, head).  X layout: [B*T, heads*128].
// ---------------------------------------------------------------------------
__global__ __launch_bounds__(128)
void norm_rope(const float* __restrict__ X, __bf16* __restrict__ Y,
               const float* __restrict__ w, int heads, int T, float outScale) {
  int bid  = blockIdx.x;
  int head = bid % heads;
  int tok  = bid / heads;          // b*T + t
  int t    = tok % T;
  int d    = threadIdx.x;          // 0..127
  const float* xp = X + ((size_t)tok * heads + head) * 128;
  float x = xp[d];

  __shared__ float red[128];
  red[d] = x * x;
  __syncthreads();
  #pragma unroll
  for (int s = 64; s > 0; s >>= 1) {
    if (d < s) red[d] += red[d + s];
    __syncthreads();
  }
  float rms = rsqrtf(red[0] / 128.0f + 1e-6f);
  float y   = x * rms * (1.0f + w[d]);

  __syncthreads();
  red[d] = y;
  __syncthreads();
  float rot = (d < 64) ? -red[d + 64] : red[d - 64];
  int   fi  = (d < 64) ? d : d - 64;
  float invf = powf(10000.0f, -(float)fi / 64.0f);
  float ang  = (float)t * invf;
  float o = y * cosf(ang) + rot * sinf(ang);
  Y[((size_t)tok * heads + head) * 128 + d] = f2bf(o * outScale);
}

// ---------------------------------------------------------------------------
// Sliding-window flash attention.  One wave per (b, head, 16-query tile).
// Key blocks of 32 -> 8 WMMA for S, 8 WMMA for PV.  Online softmax in fp32
// with width-16 shuffle reductions (C-layout keeps a row in one 16-lane half).
// Q/K/V bf16: Q [B*T, 16*128], K/V [B*T, 4*128].  Ctx bf16 [B*T, 16*128].
// ---------------------------------------------------------------------------
__global__ __launch_bounds__(32)
void attn_swa(const __bf16* __restrict__ Qm, const __bf16* __restrict__ Km,
              const __bf16* __restrict__ Vm, __bf16* __restrict__ Ctx, int T) {
  const int H = 16, Hkv = 4, Dh = 128, WIN = 512;
  const int q0 = blockIdx.x * 16;
  const int h  = blockIdx.y;
  const int b  = blockIdx.z;
  const int hk = h >> 2;                       // h / (H/Hkv)

  const int lane  = threadIdx.x;
  const int l16   = lane & 15;
  const int aoff  = (lane < 16) ? 0 : 16;
  const int koff  = (lane < 16) ? 0 : 16;
  const int half8 = (lane < 16) ? 0 : 8;

  __shared__ __bf16 P[16][32];
  __shared__ __bf16 Vs[128][32];               // transposed V tile [dim][key]

  // Q A-fragments: 4 chunks of 32 along Dh
  v16bf qf[4];
  {
    const __bf16* qrow = Qm + ((size_t)(b * T + q0 + l16) * H + h) * Dh;
    #pragma unroll
    for (int c = 0; c < 4; ++c)
      qf[c] = *(const v16bf*)(qrow + c * 32 + aoff);
  }

  const v8f vzero = {0.f,0.f,0.f,0.f,0.f,0.f,0.f,0.f};
  v8f o8[8];
  #pragma unroll
  for (int i = 0; i < 8; ++i) o8[i] = vzero;
  float m8[8], l8[8];
  #pragma unroll
  for (int i = 0; i < 8; ++i) { m8[i] = -3.0e38f; l8[i] = 0.0f; }

  int kstart = q0 - WIN; if (kstart < 0) kstart = 0;
  const int kend = q0 + 15;

  for (int kb = kstart; kb <= kend; kb += 32) {
    // ---- S = Q x K^T over this 32-key block ----
    v8f s[2] = { vzero, vzero };
    #pragma unroll
    for (int nt = 0; nt < 2; ++nt) {
      int key = kb + nt * 16 + l16;
      int keyc = (key < T) ? key : T - 1;      // masked below if out of window
      const __bf16* krow = Km + ((size_t)(b * T + keyc) * Hkv + hk) * Dh;
      #pragma unroll
      for (int c = 0; c < 4; ++c) {
        v16bf kf = *(const v16bf*)(krow + c * 32 + koff);
        s[nt] = __builtin_amdgcn_wmma_f32_16x16x32_bf16(
            false, qf[c], false, kf, (short)0, s[nt], false, false);
      }
    }

    // ---- mask + row max ----
    float rowmax[8];
    #pragma unroll
    for (int v = 0; v < 8; ++v) {
      int qpos = q0 + v + half8;
      #pragma unroll
      for (int nt = 0; nt < 2; ++nt) {
        int key  = kb + nt * 16 + l16;
        int dist = qpos - key;
        if (dist < 0 || dist > WIN) s[nt][v] = -3.0e38f;
      }
      float rm = fmaxf(s[0][v], s[1][v]);
      #pragma unroll
      for (int msk = 8; msk >= 1; msk >>= 1)
        rm = fmaxf(rm, __shfl_xor(rm, msk, 16));
      rowmax[v] = rm;
    }

    // ---- online softmax update, write P tile (bf16) ----
    #pragma unroll
    for (int v = 0; v < 8; ++v) {
      float nm = fmaxf(m8[v], rowmax[v]);
      float sc = (m8[v] == nm) ? 1.0f : __expf(m8[v] - nm);
      m8[v] = nm;
      float rs = 0.f;
      #pragma unroll
      for (int nt = 0; nt < 2; ++nt) {
        float sv = s[nt][v];
        float p  = (sv <= -1.0e37f) ? 0.0f : __expf(sv - nm);
        rs += p;
        P[v + half8][nt * 16 + l16] = f2bf(p);
      }
      #pragma unroll
      for (int msk = 8; msk >= 1; msk >>= 1)
        rs += __shfl_xor(rs, msk, 16);
      l8[v] = l8[v] * sc + rs;
      #pragma unroll
      for (int t8 = 0; t8 < 8; ++t8) o8[t8][v] *= sc;
    }

    // ---- stage V tile transposed into LDS ----
    #pragma unroll
    for (int i = 0; i < 16; ++i) {
      int chunk = lane + 32 * i;               // 0..511
      int key   = chunk >> 4;
      int dc    = (chunk & 15) * 8;
      int keyg  = kb + key; if (keyg > T - 1) keyg = T - 1;
      v8bf tmp = *(const v8bf*)(Vm + ((size_t)(b * T + keyg) * Hkv + hk) * Dh + dc);
      #pragma unroll
      for (int j = 0; j < 8; ++j) Vs[dc + j][key] = tmp[j];
    }

    // ---- O += P x V (LDS in-order within the single wave) ----
    v16bf pf = *(const v16bf*)(&P[l16][0] + aoff);
    #pragma unroll
    for (int t8 = 0; t8 < 8; ++t8) {
      v16bf vf = *(const v16bf*)&Vs[t8 * 16 + l16][koff];
      o8[t8] = __builtin_amdgcn_wmma_f32_16x16x32_bf16(
          false, pf, false, vf, (short)0, o8[t8], false, false);
    }
  }

  // ---- finalize: O / l, store bf16 ctx [B*T, H*Dh] ----
  #pragma unroll
  for (int v = 0; v < 8; ++v) {
    float inv = 1.0f / l8[v];
    int qrow = q0 + v + half8;
    __bf16* crow = Ctx + ((size_t)(b * T + qrow) * H + h) * Dh;
    #pragma unroll
    for (int t8 = 0; t8 < 8; ++t8)
      crow[t8 * 16 + l16] = f2bf(o8[t8][v] * inv);
  }
}

// ---------------------------------------------------------------------------
// host orchestration
// ---------------------------------------------------------------------------
extern "C" void kernel_launch(void* const* d_in, const int* in_sizes, int n_in,
                              void* d_out, int out_size, void* d_ws, size_t ws_size,
                              hipStream_t stream) {
  (void)in_sizes; (void)n_in; (void)out_size; (void)ws_size;
  const int B = 2, T = 2048, D = 2048, H = 16, Hkv = 4, Dh = 128;
  const int M = B * T;                         // 4096 tokens
  const int Nq = H * Dh, Nkv = Hkv * Dh;       // 2048, 512

  const float* x  = (const float*)d_in[0];
  const float* Wq = (const float*)d_in[1];
  const float* Wk = (const float*)d_in[2];
  const float* Wv = (const float*)d_in[3];
  const float* Wo = (const float*)d_in[4];
  const float* qn = (const float*)d_in[5];
  const float* kn = (const float*)d_in[6];
  float* out = (float*)d_out;

  char* ws = (char*)d_ws;
  size_t off = 0;
  auto alloc = [&](size_t bytes) -> void* {
    void* p = ws + off;
    off = (off + bytes + 255) & ~(size_t)255;
    return p;
  };
  __bf16* xb  = (__bf16*)alloc((size_t)M * D * 2);
  __bf16* WqT = (__bf16*)alloc((size_t)D * Nq * 2);
  __bf16* WkT = (__bf16*)alloc((size_t)D * Nkv * 2);
  __bf16* WvT = (__bf16*)alloc((size_t)D * Nkv * 2);
  __bf16* WoT = (__bf16*)alloc((size_t)Nq * D * 2);
  float*  Qf  = (float*)alloc((size_t)M * Nq * 4);   // reused as ctx below
  float*  Kf  = (float*)alloc((size_t)M * Nkv * 4);
  float*  Vf  = (float*)alloc((size_t)M * Nkv * 4);
  __bf16* Qb  = (__bf16*)alloc((size_t)M * Nq * 2);
  __bf16* Kb  = (__bf16*)alloc((size_t)M * Nkv * 2);
  __bf16* Vb  = (__bf16*)alloc((size_t)M * Nkv * 2);
  __bf16* Ctx = (__bf16*)Qf;   // Qf (32MB) dead after norm; ctx needs 16MB

  // 1. precision conversion / weight transposition
  cvt_f32_bf16<<<(M * D + 255) / 256, 256, 0, stream>>>(x, xb, M * D);
  transpose_cvt<<<dim3(D / 32, Nq / 32), 256, 0, stream>>>(Wq, WqT, Nq, D);
  transpose_cvt<<<dim3(D / 32, Nkv / 32), 256, 0, stream>>>(Wk, WkT, Nkv, D);
  transpose_cvt<<<dim3(D / 32, Nkv / 32), 256, 0, stream>>>(Wv, WvT, Nkv, D);
  transpose_cvt<<<dim3(Nq / 32, D / 32), 256, 0, stream>>>(Wo, WoT, D, Nq);

  // 2. QKV projections (WMMA bf16 -> f32)
  gemm_bf16_f32<<<dim3(Nq / BN, M / BM), 256, 0, stream>>>(xb, WqT, Qf, M, Nq, D);
  gemm_bf16_f32<<<dim3(Nkv / BN, M / BM), 256, 0, stream>>>(xb, WkT, Kf, M, Nkv, D);
  gemm_bf16_f32<<<dim3(Nkv / BN, M / BM), 256, 0, stream>>>(xb, WvT, Vf, M, Nkv, D);

  // 3. RMSNorm + RoPE (+ Dh^-0.5 folded into Q); V just converted
  norm_rope<<<M * H, 128, 0, stream>>>(Qf, Qb, qn, H, T, 0.08838834764831845f);
  norm_rope<<<M * Hkv, 128, 0, stream>>>(Kf, Kb, kn, Hkv, T, 1.0f);
  cvt_f32_bf16<<<(M * Nkv + 255) / 256, 256, 0, stream>>>(Vf, Vb, M * Nkv);

  // 4. sliding-window flash attention (WMMA)
  attn_swa<<<dim3(T / 16, H, B), 32, 0, stream>>>(Qb, Kb, Vb, Ctx, T);

  // 5. output projection -> f32 d_out
  gemm_bf16_f32<<<dim3(D / BN, M / BM), 256, 0, stream>>>(Ctx, WoT, out, M, D, Nq);
}
